// iRMB_13907104105093
// MI455X (gfx1250) — matrix operand order repaired and checked by reading, not compiled, not run on hardware
//
#include <hip/hip_runtime.h>
#include <hip/hip_bf16.h>
#include <stdint.h>

// ---------------- CDNA5 WMMA types ----------------
typedef __attribute__((ext_vector_type(16))) __bf16 v16bf;
typedef __attribute__((ext_vector_type(8)))  float  v8f;

#define WS     7
#define N_PIX  49        // pixels per window
#define NPAD   64        // padded M/N for 16x16 tiles
#define C_IN   256
#define C_V    1024
#define NHEAD  8
#define DHEAD  32
#define VH     64

// ---- bf16 weight cache in workspace (elements) ----
#define W_QK_ELEMS   (512 * 256)
#define W_V_ELEMS    (1024 * 256)
#define W_PROJ_ELEMS (256 * 1024)
#define W_TOTAL      (W_QK_ELEMS + W_V_ELEMS + W_PROJ_ELEMS)   // 655360

// ---- LDS layout (bytes), phase-overlaid, total 288 KB of the 320 KB WGP LDS ----
#define OFF_A   0        // sA : 64x256 bf16 (32KB)   | later sO : 64x512 bf16 (64KB, covers A+Q)
#define OFF_Q   32768    // sQ : 64x256 bf16 (32KB)
#define OFF_K   65536    // sK : 64x256 bf16 (32KB)
#define OFF_V   98304    // sVT: 512x64 bf16 (64KB, v_attn^T) | f32 staging 256x49 | later sXA 64x512
#define OFF_VI  163840   // sVI: 64x512 bf16 (64KB, v_idle row-major)
#define OFF_P   229376   // sP : 8x64x64 bf16 (64KB)  | later sOut : 64x256 f32 (64KB)
#define SMEM_BYTES 294912

// ---------------- helpers (hardware bf16 converts) ----------------
__device__ __forceinline__ __bf16 f2bf(float f) { return (__bf16)f; }
__device__ __forceinline__ float  bf2f(__bf16 b) { return (float)b; }

__device__ __forceinline__ float rsum32(float v) {
    #pragma unroll
    for (int o = 16; o > 0; o >>= 1) v += __shfl_xor(v, o, 32);
    return v;
}
__device__ __forceinline__ float rsum16(float v) {
    #pragma unroll
    for (int o = 8; o > 0; o >>= 1) v += __shfl_xor(v, o, 32);
    return v;
}
__device__ __forceinline__ float rmax16(float v) {
    #pragma unroll
    for (int o = 8; o > 0; o >>= 1) v = fmaxf(v, __shfl_xor(v, o, 32));
    return v;
}

// gfx1250 async global->LDS copy (ASYNCcnt-tracked; LDS addr = low 32 bits of flat shared ptr)
__device__ __forceinline__ void async_g2lds_b32(void* ldsp, const float* gp) {
    uint32_t lds = (uint32_t)(uintptr_t)ldsp;
    asm volatile("global_load_async_to_lds_b32 %0, %1, off"
                 :: "v"(lds), "v"(gp) : "memory");
}
__device__ __forceinline__ void wait_asynccnt0() {
    asm volatile("s_wait_asynccnt 0" ::: "memory");
}

// A fragment 16x32 bf16: lane holds row M=lane&15; elem j -> K=kb+hl*16+j
__device__ __forceinline__ v16bf loadA(const __bf16* rowp, int kb, int hl) {
    v16bf a;
    __builtin_memcpy(&a, rowp + kb + hl * 16, 32);
    return a;
}
// B fragment 32x16 bf16, K contiguous in memory (elem t -> K = base + t); pass p = row + kb + hl*16
__device__ __forceinline__ v16bf loadB16(const __bf16* p) {
    v16bf b;
    __builtin_memcpy(&b, p, 32);
    return b;
}
__device__ __forceinline__ v8f wmma_bf16(v16bf a, v16bf b, v8f c) {
    return __builtin_amdgcn_wmma_f32_16x16x32_bf16(false, a, false, b, (short)0, c, false, false);
}

// ---------------- one-time f32 -> bf16 weight conversion into workspace ----------------
__global__ __launch_bounds__(256) void cvt_weights(const float* __restrict__ qk_w,
                                                   const float* __restrict__ v_w,
                                                   const float* __restrict__ proj_w,
                                                   __bf16* __restrict__ wbuf) {
    int i = blockIdx.x * 256 + threadIdx.x;
    if (i < W_QK_ELEMS)                         wbuf[i] = (__bf16)qk_w[i];
    else if (i < W_QK_ELEMS + W_V_ELEMS)        wbuf[i] = (__bf16)v_w[i - W_QK_ELEMS];
    else if (i < W_TOTAL)                       wbuf[i] = (__bf16)proj_w[i - W_QK_ELEMS - W_V_ELEMS];
}

// ---------------- fused iRMB window-attention block: one workgroup per 7x7 window ----------------
__global__ __launch_bounds__(256) void irmb_fused(
    const float* __restrict__ x,      const float* __restrict__ pre_g,  const float* __restrict__ pre_b,
    const float* __restrict__ post_g, const float* __restrict__ post_b,
    const float* __restrict__ qk_b,   const float* __restrict__ v_b,
    const float* __restrict__ proj_b, const float* __restrict__ rpb,
    const __bf16* __restrict__ wbuf,  float* __restrict__ out)
{
    extern __shared__ char smem[];
    __bf16* sA  = (__bf16*)(smem + OFF_A);    // LN'd activations [64][256]
    __bf16* sQ  = (__bf16*)(smem + OFF_Q);    // q [64][256]
    __bf16* sK  = (__bf16*)(smem + OFF_K);    // k [64][256]
    __bf16* sVT = (__bf16*)(smem + OFF_V);    // v_attn transposed [512][64]
    __bf16* sVI = (__bf16*)(smem + OFF_VI);   // v_idle row-major [64][512]
    __bf16* sP  = (__bf16*)(smem + OFF_P);    // softmax probs [8][64][64]
    __bf16* sO  = (__bf16*)(smem + OFF_A);    // attn out [64][512]      (overlays sA+sQ)
    __bf16* sXA = (__bf16*)(smem + OFF_V);    // postLN(o) [64][512]     (overlays sVT)
    float*  sStage = (float*)(smem + OFF_V);  // raw x window [256][49]  (overlays sVT)
    float*  sOut   = (float*)(smem + OFF_P);  // proj out [64][256]      (overlays sP)

    const __bf16* qkB   = wbuf;
    const __bf16* vB    = wbuf + W_QK_ELEMS;
    const __bf16* projB = wbuf + W_QK_ELEMS + W_V_ELEMS;

    const int tid  = threadIdx.x;
    const int lane = tid & 31;
    const int hl   = (lane >> 4) & 1;   // lane half (wave32)
    const int col  = lane & 15;
    // wave id is uniform within the wave; make it explicitly scalar for SALU branching
    const int wave = __builtin_amdgcn_readfirstlane(tid >> 5);

    const int win   = blockIdx.x;       // 0..2047
    const int batch = win >> 6;
    const int w2d   = win & 63;         // window position within image (8x8)
    const int i1 = w2d >> 3, i2 = w2d & 7;
    const int h0 = i1 * WS, w0 = i2 * WS;

    const v8f vzero = {0.f, 0.f, 0.f, 0.f, 0.f, 0.f, 0.f, 0.f};

    // ---- Phase 0: async-stage window of x into LDS (channel-major), ASYNCcnt-tracked ----
    for (int idx = tid; idx < C_IN * N_PIX; idx += 256) {
        int ch = idx / N_PIX; int p = idx - ch * N_PIX;
        int r = p / WS, c = p - r * WS;
        const float* gp = x + (((size_t)batch * C_IN + ch) * 56 + (h0 + r)) * 56 + (w0 + c);
        async_g2lds_b32(sStage + ch * N_PIX + p, gp);
    }
    wait_asynccnt0();
    __syncthreads();

    // ---- Phase 1: per-pixel channel LayerNorm -> sA bf16 ----
    for (int p = wave; p < N_PIX; p += 8) {
        float s = 0.f, s2 = 0.f, vals[8];
        #pragma unroll
        for (int t = 0; t < 8; ++t) {
            float v = sStage[(lane + 32 * t) * N_PIX + p];
            vals[t] = v; s += v; s2 += v * v;
        }
        s = rsum32(s); s2 = rsum32(s2);
        float mean = s * (1.f / 256.f);
        float var  = s2 * (1.f / 256.f) - mean * mean;
        float rs   = rsqrtf(var + 1e-5f);
        #pragma unroll
        for (int t = 0; t < 8; ++t) {
            int ch = lane + 32 * t;
            sA[p * C_IN + ch] = f2bf((vals[t] - mean) * rs * pre_g[ch] + pre_b[ch]);
        }
    }
    __syncthreads();

    // ---- Phase 2: QKV GEMM  Y[p,o] = sum_c sA[p,c] * W[o,c]  (O = 512 qk + 1024 v) ----
    // tile index is wave-uniform -> scalar branches, unconditional padded-row stores
    for (int t = wave; t < 4 * 96; t += 8) {
        const int tu = __builtin_amdgcn_readfirstlane(t);
        const int mt = tu & 3, nt = tu >> 2;
        const int og = nt * 16 + col;
        const __bf16* wrow = (nt < 32) ? (qkB + (size_t)og * C_IN)
                                       : (vB + (size_t)(og - 512) * C_IN);
        const __bf16* arow = sA + (mt * 16 + col) * C_IN;
        __builtin_prefetch(wrow, 0, 1);
        v8f acc = vzero;
        #pragma unroll
        for (int kk = 0; kk < 8; ++kk) {
            v16bf a = loadA(arow, kk * 32, hl);
            v16bf b = loadB16(wrow + kk * 32 + hl * 16);
            acc = wmma_bf16(a, b, acc);
        }
        const int prow = mt * 16 + 8 * hl;       // element r stores row prow + r (< 64 always)
        if (nt < 16) {                           // ---- q ----
            float bias = qk_b[og];
            __bf16* dst = sQ + prow * C_IN + og;
            #pragma unroll
            for (int r = 0; r < 8; ++r) dst[r * C_IN] = f2bf(acc[r] + bias);
        } else if (nt < 32) {                    // ---- k ----
            float bias = qk_b[og];
            __bf16* dst = sK + prow * C_IN + (og - 256);
            #pragma unroll
            for (int r = 0; r < 8; ++r) dst[r * C_IN] = f2bf(acc[r] + bias);
        } else if (nt < 64) {                    // ---- v_attn (transposed [ch][pixel]) ----
            float bias = v_b[og - 512];
            __bf16* dst = sVT + (og - 512) * NPAD + prow;
            #pragma unroll
            for (int r = 0; r < 8; ++r) dst[r] = f2bf(acc[r] + bias);
        } else {                                 // ---- v_idle (row-major) ----
            float bias = v_b[og - 512];
            __bf16* dst = sVI + prow * 512 + (og - 1024);
            #pragma unroll
            for (int r = 0; r < 8; ++r) dst[r * 512] = f2bf(acc[r] + bias);
        }
    }
    __syncthreads();

    // re-zero sVT pad pixels m=49..63 (K-dim safety for probs x v); concurrent with softmax,
    // disjoint from sQ/sK/sP, so no extra barrier needed before the phase-4 barrier.
    for (int idx = tid; idx < 512 * (NPAD - N_PIX); idx += 256) {
        int ch = idx / (NPAD - N_PIX); int m = N_PIX + (idx - ch * (NPAD - N_PIX));
        sVT[ch * NPAD + m] = f2bf(0.f);
    }

    // ---- Phase 3: scores = qk^T/sqrt(d) + rpb, masked softmax -> sP (wave hd owns head hd) ----
    {
        const int hd = wave;                    // scalar
        const float scale = 0.17677669529663687f;
        const float* rpbw = rpb + (size_t)w2d * 169 * NHEAD + hd;
        for (int s = 0; s < 4; ++s) {
            v16bf a = loadA(sQ + (s * 16 + col) * C_IN + hd * DHEAD, 0, hl);
            v8f acc[4];
            #pragma unroll
            for (int j = 0; j < 4; ++j) {
                v16bf b = loadB16(sK + (j * 16 + col) * C_IN + hd * DHEAD + hl * 16);
                acc[j] = wmma_bf16(a, b, vzero);
            }
            #pragma unroll
            for (int r = 0; r < 8; ++r) {
                int n = s * 16 + r + 8 * hl;
                if (n < N_PIX) {
                    int rn = n / 7, cn = n - (n / 7) * 7;
                    float vals[4];
                    #pragma unroll
                    for (int j = 0; j < 4; ++j) {
                        int mm = j * 16 + col;
                        if (mm < N_PIX) {
                            int rm = mm / 7, cm = mm - rm * 7;
                            int ridx = (rn - rm + 6) * 13 + (cn - cm + 6);
                            vals[j] = acc[j][r] * scale + rpbw[(size_t)ridx * NHEAD];
                        } else vals[j] = -3.0e38f;
                    }
                    float mx = fmaxf(fmaxf(vals[0], vals[1]), fmaxf(vals[2], vals[3]));
                    mx = rmax16(mx);
                    float e0 = __expf(vals[0] - mx), e1 = __expf(vals[1] - mx);
                    float e2 = __expf(vals[2] - mx), e3 = __expf(vals[3] - mx);
                    float sum = rsum16(e0 + e1 + e2 + e3);
                    float inv = 1.f / sum;
                    sP[hd * 4096 + n * 64 +  0 + col] = f2bf(e0 * inv);
                    sP[hd * 4096 + n * 64 + 16 + col] = f2bf(e1 * inv);
                    sP[hd * 4096 + n * 64 + 32 + col] = f2bf(e2 * inv);
                    sP[hd * 4096 + n * 64 + 48 + col] = f2bf(e3 * inv);
                } else {
                    #pragma unroll
                    for (int j = 0; j < 4; ++j)
                        sP[hd * 4096 + n * 64 + j * 16 + col] = f2bf(0.f);
                }
            }
        }
    }
    __syncthreads();

    // ---- Phase 4: o = probs x v_attn  (B from transposed sVT, K contiguous) -> sO bf16 ----
    for (int t = wave; t < 128; t += 8) {
        const int tu = __builtin_amdgcn_readfirstlane(t);
        const int hd = tu >> 4, mt = tu & 3, nt = (tu >> 2) & 3;
        const __bf16* arow = sP + hd * 4096 + (mt * 16 + col) * 64;
        const int ch = hd * VH + nt * 16 + col;
        v8f acc = vzero;
        #pragma unroll
        for (int kk = 0; kk < 2; ++kk) {
            v16bf a = loadA(arow, kk * 32, hl);
            v16bf b = loadB16(sVT + ch * NPAD + kk * 32 + hl * 16);
            acc = wmma_bf16(a, b, acc);
        }
        __bf16* dst = sO + (mt * 16 + 8 * hl) * 512 + ch;
        #pragma unroll
        for (int r = 0; r < 8; ++r) dst[r * 512] = f2bf(acc[r]);
    }
    __syncthreads();

    // ---- Phase 5: post-LN over the 512 attention channels -> sXA (overlays dead sVT) ----
    for (int p = wave; p < N_PIX; p += 8) {
        float s = 0.f, s2 = 0.f, vals[16];
        #pragma unroll
        for (int t = 0; t < 16; ++t) {
            float v = bf2f(sO[p * 512 + lane + 32 * t]);
            vals[t] = v; s += v; s2 += v * v;
        }
        s = rsum32(s); s2 = rsum32(s2);
        float mean = s * (1.f / 512.f);
        float var  = s2 * (1.f / 512.f) - mean * mean;
        float rs   = rsqrtf(var + 1e-5f);
        #pragma unroll
        for (int t = 0; t < 16; ++t) {
            int ch = lane + 32 * t;
            sXA[p * 512 + ch] = f2bf((vals[t] - mean) * rs * post_g[ch] + post_b[ch]);
        }
    }
    __syncthreads();

    // ---- Phase 6: exact GELU on xcat = [sXA | sVI] (in place) ----
    for (int idx = tid; idx < N_PIX * 512; idx += 256) {
        float f = bf2f(sXA[idx]);
        sXA[idx] = f2bf(0.5f * f * (1.f + erff(f * 0.70710678f)));
    }
    for (int idx = tid; idx < N_PIX * 512; idx += 256) {
        float f = bf2f(sVI[idx]);
        sVI[idx] = f2bf(0.5f * f * (1.f + erff(f * 0.70710678f)));
    }
    __syncthreads();

    // ---- Phase 7: projection GEMM  out[p,oc] = sum_c xm[p,c] * proj_w[oc,c] ----
    for (int t = wave; t < 64; t += 8) {
        const int tu = __builtin_amdgcn_readfirstlane(t);
        const int mt = tu & 3, nt = tu >> 2;
        const int oc = nt * 16 + col;
        const __bf16* wrow = projB + (size_t)oc * C_V;
        __builtin_prefetch(wrow, 0, 1);
        v8f acc = vzero;
        #pragma unroll
        for (int kk = 0; kk < 32; ++kk) {
            const __bf16* arow = (kk < 16) ? (sXA + (mt * 16 + col) * 512)
                                           : (sVI + (mt * 16 + col) * 512);
            int kb = (kk < 16) ? kk * 32 : (kk - 16) * 32;
            v16bf a = loadA(arow, kb, hl);
            v16bf b = loadB16(wrow + kk * 32 + hl * 16);
            acc = wmma_bf16(a, b, acc);
        }
        float pb = proj_b[oc];
        float* dst = sOut + (mt * 16 + 8 * hl) * 256 + oc;
        #pragma unroll
        for (int r = 0; r < 8; ++r) dst[r * 256] = acc[r] + pb;   // pad rows written too (ignored)
    }
    __syncthreads();

    // ---- Phase 8: residual add + roll(2,2) writeout ----
    for (int idx = tid; idx < 256 * N_PIX; idx += 256) {
        int oc = idx / N_PIX; int p = idx - oc * N_PIX;
        int r = p / WS, c = p - r * WS;
        int hh = h0 + r, ww = w0 + c;
        int ho = hh + 2; if (ho >= 56) ho -= 56;
        int wo = ww + 2; if (wo >= 56) wo -= 56;
        size_t ibase = ((size_t)batch * 256 + oc) * 56;
        out[(ibase + ho) * 56 + wo] = x[(ibase + hh) * 56 + ww] + sOut[p * 256 + oc];
    }
}

extern "C" void kernel_launch(void* const* d_in, const int* in_sizes, int n_in,
                              void* d_out, int out_size, void* d_ws, size_t ws_size,
                              hipStream_t stream) {
    const float* x      = (const float*)d_in[0];
    const float* pre_g  = (const float*)d_in[1];
    const float* pre_b  = (const float*)d_in[2];
    const float* post_g = (const float*)d_in[3];
    const float* post_b = (const float*)d_in[4];
    const float* qk_w   = (const float*)d_in[5];
    const float* qk_b   = (const float*)d_in[6];
    const float* v_w    = (const float*)d_in[7];
    const float* v_b    = (const float*)d_in[8];
    const float* proj_w = (const float*)d_in[9];
    const float* proj_b = (const float*)d_in[10];
    const float* rpb    = (const float*)d_in[11];
    float* out          = (float*)d_out;
    __bf16* wbuf        = (__bf16*)d_ws;     // 1.25 MB bf16 weight cache (L2-resident)

    // one-time (per launch) weight downconvert; cheap (2.6MB read) and deterministic
    cvt_weights<<<(W_TOTAL + 255) / 256, 256, 0, stream>>>(qk_w, v_w, proj_w, wbuf);

    // opt in to >64KB dynamic LDS (CDNA5 WGP has 320KB); capture-safe
    (void)hipFuncSetAttribute((const void*)irmb_fused,
                              hipFuncAttributeMaxDynamicSharedMemorySize, SMEM_BYTES);

    irmb_fused<<<2048, 256, SMEM_BYTES, stream>>>(
        x, pre_g, pre_b, post_g, post_b, qk_b, v_b, proj_b, rpb, wbuf, out);
}
